// AgentAttention_9113920602397
// MI455X (gfx1250) — matrix operand order repaired and checked by reading, not compiled, not run on hardware
//
#include <hip/hip_runtime.h>
#include <hip/hip_bf16.h>
#include <stdint.h>

typedef __attribute__((ext_vector_type(16))) _Float16 v16h;
typedef __attribute__((ext_vector_type(8)))  float    v8f;
typedef __attribute__((ext_vector_type(4)))  int      v4i;

#define B_  16
#define C_  256
#define H_  56
#define W_  56
#define N_  3136
#define NH_ 8
#define HD_ 32
#define AG_ 49
#define AGP 64   // agents padded to multiple of 16
#define SCALE_ 0.17677669529663687f  // 32^-0.5

#define LDS_LDA 40   // padded row stride (halfs) for 32-wide K tiles: kills bank conflicts

// ---------------- CDNA5 async global->LDS copy (ASYNCcnt path) ----------------
typedef __attribute__((address_space(1))) v4i as1_v4i;
typedef __attribute__((address_space(3))) v4i as3_v4i;

static __device__ __forceinline__ void async_copy16(const void* g, void* l) {
#if defined(__has_builtin) && __has_builtin(__builtin_amdgcn_global_load_async_to_lds_b128)
  __builtin_amdgcn_global_load_async_to_lds_b128(
      (as1_v4i*)(uintptr_t)g,
      (as3_v4i*)(uintptr_t)(uint32_t)(uintptr_t)l, 0, 0);
#else
  *(int4*)l = *(const int4*)g;
#endif
}
static __device__ __forceinline__ void async_wait0() {
#if defined(__has_builtin) && __has_builtin(__builtin_amdgcn_global_load_async_to_lds_b128)
#if __has_builtin(__builtin_amdgcn_s_wait_asynccnt)
  __builtin_amdgcn_s_wait_asynccnt(0);
#else
  asm volatile("s_wait_asynccnt 0x0" ::: "memory");
#endif
#endif
}

// ---------------- WMMA fragment loaders (wave32, 16x16x32 f16) ----------------
// A: 16x32 tile, memory row-major [M][K], lane=M (mod 16), lane-half selects K-phase.
static __device__ __forceinline__ v16h load_a_frag(const _Float16* A, int lda, int lane) {
  const int m  = lane & 15;
  const int hl = lane >> 4;
  const _Float16* base = A + (size_t)m * lda + (hl << 3);
  v16h r;
#pragma unroll
  for (int g = 0; g < 2; ++g)
#pragma unroll
    for (int e = 0; e < 8; ++e)
      r[g * 8 + e] = base[g * 16 + e];   // K = g*16 + half*8 + e
  return r;
}
// B: 32x16 tile stored transposed ("BT" row-major [Ncol][K]), lane=N, K = (e/8)*16 + half*8 + e%8.
static __device__ __forceinline__ v16h load_b_frag(const _Float16* BT, int ldb, int lane) {
  const int n  = lane & 15;
  const int hl = lane >> 4;
  const _Float16* base = BT + (size_t)n * ldb + (hl << 3);
  v16h r;
#pragma unroll
  for (int e = 0; e < 16; ++e) r[e] = base[(e >> 3) * 16 + (e & 7)];
  return r;
}

// ---------------- prep: f32 -> f16 conversions ----------------
__global__ void cvt_f16_kernel(const float* src, _Float16* dst, int n) {
  int i = blockIdx.x * blockDim.x + threadIdx.x;
  if (i < n) dst[i] = (_Float16)src[i];
}

// x [B,C,N] f32 -> xT [B,N,C] f16
__global__ void transpose_x_kernel(const float* x, _Float16* xT) {
  int i = blockIdx.x * blockDim.x + threadIdx.x;
  if (i >= B_ * N_ * C_) return;
  int c = i & (C_ - 1);
  int bn = i >> 8;
  int n = bn % N_;
  int b = bn / N_;
  xT[i] = (_Float16)x[((size_t)(b * C_ + c)) * N_ + n];
}

// adaptive avg pool 56x56 -> 7x7; agent f16 [B,nh,AGP,HD], rows >= 49 zeroed
__global__ void pool_kernel(const float* x, _Float16* agent) {
  int i = blockIdx.x * blockDim.x + threadIdx.x;
  if (i >= B_ * NH_ * AGP * HD_) return;
  int d  = i & (HD_ - 1);
  int a  = (i >> 5) & (AGP - 1);
  int bh = i >> 11;
  int h  = bh & (NH_ - 1);
  int b  = bh >> 3;
  _Float16 val = (_Float16)0.0f;
  if (a < AG_) {
    int c = h * HD_ + d;
    int py = a / 7, px = a % 7;
    const float* p = x + ((size_t)(b * C_ + c)) * N_;
    float s = 0.0f;
#pragma unroll
    for (int yy = 0; yy < 8; ++yy)
#pragma unroll
      for (int xx = 0; xx < 8; ++xx)
        s += p[(py * 8 + yy) * W_ + px * 8 + xx];
    val = (_Float16)(s * (1.0f / 64.0f));
  }
  agent[i] = val;
}

// ================= LDS-tiled GEMM mainloop (64x64 block tile, K=256) =================
// 128 threads = 4 waves. Wave w owns M-strip w*16; computes 4 N-tiles (acc[0..3]).
// A [M x 256] row-major, BT [Ncols x 256] row-major. Async global->LDS staging.
static __device__ __forceinline__ void gemm_block_64x64(
    const _Float16* Ag, const _Float16* Bg,   // tile bases: A + m0*256, BT + j0*256
    _Float16* As, _Float16* Bs,               // LDS: 64*LDS_LDA halfs each
    int wave, int lane, v8f acc[4]) {
  for (int kk = 0; kk < C_; kk += 32) {
#pragma unroll
    for (int rep = 0; rep < 2; ++rep) {
      int chunk = threadIdx.x + rep * 128;     // 256 chunks of 8 halfs per tile
      int row = chunk >> 2, c4 = chunk & 3;
      async_copy16(Ag + (size_t)row * C_ + kk + c4 * 8, &As[row * LDS_LDA + c4 * 8]);
      async_copy16(Bg + (size_t)row * C_ + kk + c4 * 8, &Bs[row * LDS_LDA + c4 * 8]);
    }
    async_wait0();
    __syncthreads();
    v16h af = load_a_frag(&As[(wave * 16) * LDS_LDA], LDS_LDA, lane);
#pragma unroll
    for (int jn = 0; jn < 4; ++jn) {
      v16h bf = load_b_frag(&Bs[(jn * 16) * LDS_LDA], LDS_LDA, lane);
      acc[jn] = __builtin_amdgcn_wmma_f32_16x16x32_f16(false, af, false, bf, (short)0,
                                                       acc[jn], false, false);
    }
    __syncthreads();
  }
}

// ---------------- QKV GEMM: [B*N,256] x [256,768], LDS-tiled ----------------
__global__ void gemm_qkv_kernel(const _Float16* xT, const _Float16* wqkv,
                                _Float16* q, _Float16* k, _Float16* vT) {
  __shared__ _Float16 As[64 * LDS_LDA];
  __shared__ _Float16 Bs[64 * LDS_LDA];
  const int wave = threadIdx.x >> 5;
  const int lane = threadIdx.x & 31;
  const int bm = blockIdx.x / 12, bn = blockIdx.x % 12;   // 784 x 12 blocks
  const int m0 = bm * 64, j0 = bn * 64;
  v8f acc[4] = {};
  gemm_block_64x64(xT + (size_t)m0 * C_, wqkv + (size_t)j0 * C_, As, Bs, wave, lane, acc);
  const int col = lane & 15, hl = lane >> 4;
#pragma unroll
  for (int jn = 0; jn < 4; ++jn) {
    const int j = j0 + jn * 16 + col;
#pragma unroll
    for (int e = 0; e < 8; ++e) {
      const int m  = m0 + wave * 16 + hl * 8 + e;
      const int bb = m / N_, n = m % N_;
      _Float16 val = (_Float16)acc[jn][e];
      if (j < C_) {
        int h = j >> 5, d = j & 31;
        q[(((size_t)(bb * NH_ + h)) * N_ + n) * HD_ + d] = val;
      } else if (j < 2 * C_) {
        int jj = j - C_;
        int h = jj >> 5, d = jj & 31;
        k[(((size_t)(bb * NH_ + h)) * N_ + n) * HD_ + d] = val;
      } else {
        int c = j - 2 * C_;
        vT[((size_t)(bb * C_ + c)) * N_ + n] = val;  // v as image [B,C,N]
      }
    }
  }
}

// ---------------- proj GEMM: [B*N,256] x [256,256], LDS-tiled, fused epilogue ----------------
__global__ void gemm_proj_kernel(const _Float16* z, const _Float16* wproj,
                                 const float* proj_b, const float* x, float* out) {
  __shared__ _Float16 As[64 * LDS_LDA];
  __shared__ _Float16 Bs[64 * LDS_LDA];
  const int wave = threadIdx.x >> 5;
  const int lane = threadIdx.x & 31;
  const int bm = blockIdx.x >> 2, bn = blockIdx.x & 3;    // 784 x 4 blocks
  const int m0 = bm * 64, j0 = bn * 64;
  v8f acc[4] = {};
  gemm_block_64x64(z + (size_t)m0 * C_, wproj + (size_t)j0 * C_, As, Bs, wave, lane, acc);
  const int col = lane & 15, hl = lane >> 4;
#pragma unroll
  for (int jn = 0; jn < 4; ++jn) {
    const int j = j0 + jn * 16 + col;
    const float pb = proj_b[j];
#pragma unroll
    for (int e = 0; e < 8; ++e) {
      const int m = m0 + wave * 16 + hl * 8 + e;
      const int bb = m / N_, n = m % N_;
      const size_t o = ((size_t)(bb * C_ + j)) * N_ + n;
      out[o] = x[o] + acc[jn][e] + pb;
    }
  }
}

// ---------------- stage1 scores: agent[64,32] x k^T -> S[64,N] per (b,h) ----------------
__global__ void gemm_s1_kernel(const _Float16* agent, const _Float16* kb, float* S) {
  const int wave = (blockIdx.x * blockDim.x + threadIdx.x) >> 5;
  const int lane = threadIdx.x & 31;
  const int total = B_ * NH_ * (AGP / 16) * (N_ / 16);  // 128*784
  if (wave >= total) return;
  const int bh = wave / 784, t = wave % 784;
  const int tm = t / 196, tn = t % 196;
  const _Float16* A  = agent + (size_t)bh * AGP * HD_ + tm * 16 * HD_;
  const _Float16* BT = kb + ((size_t)bh * N_ + tn * 16) * HD_;
  v16h af = load_a_frag(A, HD_, lane);
  v16h bf = load_b_frag(BT, HD_, lane);
  v8f acc = {};
  acc = __builtin_amdgcn_wmma_f32_16x16x32_f16(false, af, false, bf, (short)0, acc, false, false);
  const int col = lane & 15, hl = lane >> 4;
  float* out = S + (size_t)bh * AGP * N_;
#pragma unroll
  for (int e = 0; e < 8; ++e)
    out[(size_t)(tm * 16 + hl * 8 + e) * N_ + tn * 16 + col] = acc[e];
}

// softmax over N=3136 for rows a<49; scale folded in; write f16 attn
__global__ void softmax1_kernel(const float* S, _Float16* attn) {
  const int r = blockIdx.x;               // B*nh*49 rows
  const int bh = r / AG_, a = r % AG_;
  const float* row = S + ((size_t)bh * AGP + a) * N_;
  _Float16* orow = attn + ((size_t)bh * AGP + a) * N_;
  __shared__ float red[256];
  float mx = -1e30f;
  for (int i = threadIdx.x; i < N_; i += 256) mx = fmaxf(mx, row[i] * SCALE_);
  red[threadIdx.x] = mx; __syncthreads();
  for (int s = 128; s > 0; s >>= 1) {
    if ((int)threadIdx.x < s) red[threadIdx.x] = fmaxf(red[threadIdx.x], red[threadIdx.x + s]);
    __syncthreads();
  }
  mx = red[0]; __syncthreads();
  float sum = 0.0f;
  for (int i = threadIdx.x; i < N_; i += 256) sum += __expf(row[i] * SCALE_ - mx);
  red[threadIdx.x] = sum; __syncthreads();
  for (int s = 128; s > 0; s >>= 1) {
    if ((int)threadIdx.x < s) red[threadIdx.x] += red[threadIdx.x + s];
    __syncthreads();
  }
  const float inv = 1.0f / red[0];
  for (int i = threadIdx.x; i < N_; i += 256)
    orow[i] = (_Float16)(__expf(row[i] * SCALE_ - mx) * inv);
}

// ---------------- agent_v = attn[64,N] x v[N,32] per (b,h) ----------------
__global__ void gemm_av_kernel(const _Float16* attn, const _Float16* vT, float* agv) {
  const int wave = (blockIdx.x * blockDim.x + threadIdx.x) >> 5;
  const int lane = threadIdx.x & 31;
  const int total = B_ * NH_ * 8;   // 4 M-tiles x 2 N-tiles
  if (wave >= total) return;
  const int bh = wave >> 3, t = wave & 7;
  const int tm = t >> 1, tn = t & 1;
  const int b = bh >> 3, h = bh & (NH_ - 1);
  const _Float16* A  = attn + (size_t)bh * AGP * N_ + (size_t)tm * 16 * N_;
  const _Float16* BT = vT + ((size_t)b * C_ + h * HD_ + tn * 16) * N_;  // [d][n]
  v8f acc = {};
  for (int kk = 0; kk < N_; kk += 32) {
    v16h af = load_a_frag(A + kk, N_, lane);
    v16h bf = load_b_frag(BT + kk, N_, lane);
    acc = __builtin_amdgcn_wmma_f32_16x16x32_f16(false, af, false, bf, (short)0, acc, false, false);
  }
  const int col = lane & 15, hl = lane >> 4;
  float* out = agv + (size_t)bh * AGP * HD_;
#pragma unroll
  for (int e = 0; e < 8; ++e)
    out[(size_t)(tm * 16 + hl * 8 + e) * HD_ + tn * 16 + col] = acc[e];
}

// agent_vT [bh][d][AGP], zero pad a>=49 (BT layout for stage-2 second GEMM)
__global__ void transpose_av_kernel(const float* agv, _Float16* agvT) {
  int i = blockIdx.x * blockDim.x + threadIdx.x;
  if (i >= B_ * NH_ * HD_ * AGP) return;
  int a  = i & (AGP - 1);
  int d  = (i >> 6) & (HD_ - 1);
  int bh = i >> 11;
  agvT[i] = (a < AG_) ? (_Float16)agv[(size_t)bh * AGP * HD_ + a * HD_ + d] : (_Float16)0.0f;
}

// ---------------- stage2 scores: q[N,32] x agent^T -> S2[N,64] per (b,h) ----------------
__global__ void gemm_s2_kernel(const _Float16* qb, const _Float16* agent, float* S2) {
  const int wave = (blockIdx.x * blockDim.x + threadIdx.x) >> 5;
  const int lane = threadIdx.x & 31;
  const int total = B_ * NH_ * (N_ / 16) * (AGP / 16);  // 128*784
  if (wave >= total) return;
  const int bh = wave / 784, t = wave % 784;
  const int tm = t / 4, tn = t % 4;
  const _Float16* A  = qb + ((size_t)bh * N_ + tm * 16) * HD_;
  const _Float16* BT = agent + (size_t)bh * AGP * HD_ + tn * 16 * HD_;
  v16h af = load_a_frag(A, HD_, lane);
  v16h bf = load_b_frag(BT, HD_, lane);
  v8f acc = {};
  acc = __builtin_amdgcn_wmma_f32_16x16x32_f16(false, af, false, bf, (short)0, acc, false, false);
  const int col = lane & 15, hl = lane >> 4;
  float* out = S2 + (size_t)bh * N_ * AGP;
#pragma unroll
  for (int e = 0; e < 8; ++e)
    out[(size_t)(tm * 16 + hl * 8 + e) * AGP + tn * 16 + col] = acc[e];
}

// softmax over 49 agents per row; write f16 P padded to 64 with zeros
__global__ void softmax2_kernel(const float* S2, _Float16* P) {
  int r = blockIdx.x * blockDim.x + threadIdx.x;
  if (r >= B_ * NH_ * N_) return;
  const float* row = S2 + (size_t)r * AGP;
  _Float16* orow = P + (size_t)r * AGP;
  float mx = -1e30f;
  for (int i = 0; i < AG_; ++i) mx = fmaxf(mx, row[i] * SCALE_);
  float sum = 0.0f;
  for (int i = 0; i < AG_; ++i) sum += __expf(row[i] * SCALE_ - mx);
  const float inv = 1.0f / sum;
  for (int i = 0; i < AG_; ++i) orow[i] = (_Float16)(__expf(row[i] * SCALE_ - mx) * inv);
  for (int i = AG_; i < AGP; ++i) orow[i] = (_Float16)0.0f;
}

// ---------------- x_out = P[N,64] x agent_v[64,32] per (b,h) ----------------
__global__ void gemm_xo_kernel(const _Float16* P, const _Float16* agvT, float* xo) {
  const int wave = (blockIdx.x * blockDim.x + threadIdx.x) >> 5;
  const int lane = threadIdx.x & 31;
  const int total = B_ * NH_ * (N_ / 16) * (HD_ / 16);  // 128*392
  if (wave >= total) return;
  const int bh = wave / 392, t = wave % 392;
  const int tm = t >> 1, tn = t & 1;
  const _Float16* A  = P + ((size_t)bh * N_ + tm * 16) * AGP;
  const _Float16* BT = agvT + (size_t)bh * HD_ * AGP + tn * 16 * AGP;
  v8f acc = {};
  for (int kk = 0; kk < AGP; kk += 32) {
    v16h af = load_a_frag(A + kk, AGP, lane);
    v16h bf = load_b_frag(BT + kk, AGP, lane);
    acc = __builtin_amdgcn_wmma_f32_16x16x32_f16(false, af, false, bf, (short)0, acc, false, false);
  }
  const int col = lane & 15, hl = lane >> 4;
  float* out = xo + (size_t)bh * N_ * HD_;
#pragma unroll
  for (int e = 0; e < 8; ++e)
    out[(size_t)(tm * 16 + hl * 8 + e) * HD_ + tn * 16 + col] = acc[e];
}

// ---------------- depthwise 3x3 on v image + x_out add -> z f16 [B,N,C] ----------------
__global__ void dwc_z_kernel(const _Float16* vT, const float* dwc_w, const float* dwc_b,
                             const float* xo, _Float16* z) {
  int i = blockIdx.x * blockDim.x + threadIdx.x;
  if (i >= B_ * C_ * N_) return;
  int n = i % N_;
  int bc = i / N_;
  int c = bc & (C_ - 1);
  int b = bc >> 8;
  int y = n / W_, xc = n % W_;
  const _Float16* vp = vT + (size_t)bc * N_;
  const float* w = dwc_w + c * 9;
  float s = dwc_b[c];
#pragma unroll
  for (int dy = -1; dy <= 1; ++dy)
#pragma unroll
    for (int dx = -1; dx <= 1; ++dx) {
      int yy = y + dy, xx = xc + dx;
      if (yy >= 0 && yy < H_ && xx >= 0 && xx < W_)
        s += (float)vp[yy * W_ + xx] * w[(dy + 1) * 3 + (dx + 1)];
    }
  int h = c >> 5, d = c & 31;
  s += xo[(((size_t)(b * NH_ + h)) * N_ + n) * HD_ + d];
  z[((size_t)b * N_ + n) * C_ + c] = (_Float16)s;
}

// ---------------- launcher ----------------
extern "C" void kernel_launch(void* const* d_in, const int* in_sizes, int n_in,
                              void* d_out, int out_size, void* d_ws, size_t ws_size,
                              hipStream_t stream) {
  const float* x      = (const float*)d_in[0];
  const float* qkv_w  = (const float*)d_in[1];
  const float* proj_w = (const float*)d_in[2];
  const float* proj_b = (const float*)d_in[3];
  const float* dwc_w  = (const float*)d_in[4];
  const float* dwc_b  = (const float*)d_in[5];
  float* out = (float*)d_out;

  char* ws = (char*)d_ws;
  size_t off = 0;
  auto alloc = [&](size_t bytes) -> void* {
    off = (off + 255) & ~(size_t)255;
    void* p = ws + off;
    off += bytes;
    return p;
  };

  const size_t BNC = (size_t)B_ * N_ * C_;  // 12,845,056
  _Float16* xT    = (_Float16*)alloc(BNC * 2);                         // later reused as z
  _Float16* wqkv  = (_Float16*)alloc((size_t)3 * C_ * C_ * 2);
  _Float16* wproj = (_Float16*)alloc((size_t)C_ * C_ * 2);
  _Float16* qb    = (_Float16*)alloc(BNC * 2);
  _Float16* kb    = (_Float16*)alloc(BNC * 2);
  _Float16* vT    = (_Float16*)alloc(BNC * 2);
  _Float16* agent = (_Float16*)alloc((size_t)B_ * NH_ * AGP * HD_ * 2);
  float*    S     = (float*)alloc((size_t)B_ * NH_ * AGP * N_ * 4);    // scores1, reused scores2
  _Float16* attnP = (_Float16*)alloc((size_t)B_ * NH_ * AGP * N_ * 2); // attn, reused as P
  float*    agv   = (float*)alloc((size_t)B_ * NH_ * AGP * HD_ * 4);
  _Float16* agvT  = (_Float16*)alloc((size_t)B_ * NH_ * HD_ * AGP * 2);
  float*    xo    = (float*)alloc(BNC * 4);
  _Float16* z     = xT;  // xT dead after QKV GEMM

  const int TB = 256;
  auto cdivb = [](long n, long b) -> int { return (int)((n + b - 1) / b); };

  transpose_x_kernel<<<cdivb(BNC, TB), TB, 0, stream>>>(x, xT);
  cvt_f16_kernel<<<cdivb(3 * C_ * C_, TB), TB, 0, stream>>>(qkv_w, wqkv, 3 * C_ * C_);
  cvt_f16_kernel<<<cdivb(C_ * C_, TB), TB, 0, stream>>>(proj_w, wproj, C_ * C_);
  pool_kernel<<<cdivb(B_ * NH_ * AGP * HD_, TB), TB, 0, stream>>>(x, agent);

  // QKV: 784 M-blocks x 12 N-blocks, 128 threads (4 waves) each, LDS tiled + async copies
  gemm_qkv_kernel<<<784 * 12, 128, 0, stream>>>(xT, wqkv, qb, kb, vT);

  {  // agent . K^T
    long waves = (long)B_ * NH_ * (AGP / 16) * (N_ / 16);
    gemm_s1_kernel<<<cdivb(waves * 32, TB), TB, 0, stream>>>(agent, kb, S);
  }
  softmax1_kernel<<<B_ * NH_ * AG_, 256, 0, stream>>>(S, attnP);
  {  // attn . V
    long waves = (long)B_ * NH_ * 8;
    gemm_av_kernel<<<cdivb(waves * 32, TB), TB, 0, stream>>>(attnP, vT, agv);
  }
  transpose_av_kernel<<<cdivb(B_ * NH_ * HD_ * AGP, TB), TB, 0, stream>>>(agv, agvT);
  {  // q . agent^T
    long waves = (long)B_ * NH_ * (N_ / 16) * (AGP / 16);
    gemm_s2_kernel<<<cdivb(waves * 32, TB), TB, 0, stream>>>(qb, agent, S);
  }
  softmax2_kernel<<<cdivb(B_ * NH_ * N_, TB), TB, 0, stream>>>(S, attnP);
  {  // P . agent_v
    long waves = (long)B_ * NH_ * (N_ / 16) * (HD_ / 16);
    gemm_xo_kernel<<<cdivb(waves * 32, TB), TB, 0, stream>>>(attnP, agvT, xo);
  }
  dwc_z_kernel<<<cdivb((long)B_ * C_ * N_, TB), TB, 0, stream>>>(vT, dwc_w, dwc_b, xo, z);

  // proj + residual: 784 M-blocks x 4 N-blocks, LDS tiled + async copies
  gemm_proj_kernel<<<784 * 4, 128, 0, stream>>>(z, wproj, proj_b, x, out);
}
